// EVCBlock_20890720928164
// MI455X (gfx1250) — compile-verified, matrix operand order checked by reading, not run on hardware
//
#include <hip/hip_runtime.h>
#include <math.h>

// ---------------- CDNA5 WMMA plumbing ----------------
typedef __bf16 bf16;
typedef bf16  v16bf __attribute__((ext_vector_type(16)));
typedef float v8f   __attribute__((ext_vector_type(8)));

#define ACT_NONE 0
#define ACT_RELU 1
#define ACT_SILU 2
#define ACT_GELU 3

#define ASTR 40   // LDS row stride in ushorts: 80B rows -> 8B-aligned 4-elem groups
#define BSTR 40

__device__ __forceinline__ unsigned short f2bfu(float f) {
  unsigned u = __float_as_uint(f);
  u += 0x7FFFu + ((u >> 16) & 1u);        // round-to-nearest-even to bf16
  return (unsigned short)(u >> 16);
}

__device__ __forceinline__ void store_bf4(unsigned short* dst, float4 v) {
  union { unsigned short s[4]; uint2 u; } t;
  t.s[0] = f2bfu(v.x); t.s[1] = f2bfu(v.y);
  t.s[2] = f2bfu(v.z); t.s[3] = f2bfu(v.w);
  *(uint2*)dst = t.u;                      // 8B-aligned LDS store
}

__device__ __forceinline__ float act_apply(float v, int act) {
  if (act == ACT_RELU) return fmaxf(v, 0.f);
  if (act == ACT_SILU) return v / (1.f + expf(-v));
  if (act == ACT_GELU) return 0.5f * v * (1.f + erff(v * 0.70710678118654752f));
  return v;
}

// Fragment gather per ISA 16-bit A/B layouts: lane l holds row (l&15),
// element i -> k = (i&7) + ((i>>3)<<4) + ((l>>4)<<3)
__device__ __forceinline__ v16bf load_frag(const unsigned short* __restrict__ row,
                                           int lane) {
  union { v16bf v; unsigned short s[16]; } u;
  const int kb = (lane >> 4) << 3;
#pragma unroll
  for (int i = 0; i < 16; ++i) {
    int k = (i & 7) + ((i >> 3) << 4) + kb;
    u.s[i] = row[k];
  }
  return u.v;
}

__device__ __forceinline__ v8f wmma_bf16(v16bf a, v16bf b, v8f c) {
  return __builtin_amdgcn_wmma_f32_16x16x32_bf16(false, a, false, b,
                                                 (short)0, c, false, false);
}

// ---------------- GEMM: Y[b] = act(affine(W[Cout,Cin] x X[b][Cin,NSP]) + bias)
// Block tile MT x 64, 8 waves. MT=128: 4x2 wave grid (32x32/wave, 4 WMMA/step).
// MT=64: 2x4 wave grid (32x16/wave). All launch shapes are exact multiples,
// so staging and epilogue are unguarded (no EXEC divergence around WMMA).
template <int MT>
__global__ __launch_bounds__(256) void k_gemm(
    const float* __restrict__ Wt, const float* __restrict__ X,
    const float* __restrict__ X2, int Csplit,
    float* __restrict__ Y, int Cout, int Cin, int NSP,
    const float* __restrict__ gq, const float* __restrict__ bq,
    const float* __restrict__ bias, int act)
{
  constexpr int MW = MT / 32;           // m-waves
  constexpr int NW = 8 / MW;            // n-waves
  constexpr int NSUB = 64 / (NW * 16);  // n subtiles per wave
  __shared__ __align__(16) unsigned short As[MT * ASTR];
  __shared__ __align__(16) unsigned short Bs[64 * BSTR];
  const int tid  = threadIdx.x;
  const int lane = tid & 31, w = tid >> 5;
  const int wm = w / NW, wn = w % NW;
  const int m0 = blockIdx.y * MT;
  const int p0 = blockIdx.x * 64;
  const int b  = blockIdx.z;
  const float* Xb  = X + (size_t)b * Csplit * NSP;
  const float* X2b = X2 ? (X2 + (size_t)b * (size_t)(Cin - Csplit) * NSP) : nullptr;
  float* Yb = Y + (size_t)b * Cout * NSP;

  v8f acc[2][NSUB];
#pragma unroll
  for (int i = 0; i < 2; ++i)
#pragma unroll
    for (int j = 0; j < NSUB; ++j) acc[i][j] = (v8f){0,0,0,0,0,0,0,0};

  for (int k0 = 0; k0 < Cin; k0 += 32) {
    __syncthreads();
#pragma unroll
    for (int e = 0; e < MT / 32; ++e) {        // A: MT x 32, float4 loads
      int idx = tid + e * 256;
      int m = idx >> 3, k4 = (idx & 7) << 2;
      float4 v = *(const float4*)(Wt + (size_t)(m0 + m) * Cin + k0 + k4);
      store_bf4(&As[m * ASTR + k4], v);
    }
#pragma unroll
    for (int e = 0; e < 2; ++e) {              // B: 32 x 64, float4 along n
      int idx = tid + e * 256;
      int k = idx >> 4, n4 = (idx & 15) << 2;
      int kg = k0 + k;
      const float* src = (kg < Csplit) ? (Xb + (size_t)kg * NSP)
                                       : (X2b + (size_t)(kg - Csplit) * NSP);
      float4 v = *(const float4*)(src + p0 + n4);
      unsigned short* d = &Bs[n4 * BSTR + k];
      d[0 * BSTR] = f2bfu(v.x);
      d[1 * BSTR] = f2bfu(v.y);
      d[2 * BSTR] = f2bfu(v.z);
      d[3 * BSTR] = f2bfu(v.w);
    }
    {                                          // prefetch next K-block
      int kn = k0 + 32;
      if (kn < Cin) {
        __builtin_prefetch(Wt + (size_t)(m0 + (tid % MT)) * Cin + kn, 0, 1);
        int kg = kn + (tid & 31);
        const float* src = (kg < Csplit) ? (Xb + (size_t)kg * NSP)
                                         : (X2b + (size_t)(kg - Csplit) * NSP);
        __builtin_prefetch(src + p0 + ((tid >> 5) << 3), 0, 1);
      }
    }
    __syncthreads();
    v16bf afr[2], bfr[NSUB];
#pragma unroll
    for (int j = 0; j < NSUB; ++j)
      bfr[j] = load_frag(&Bs[((wn * NSUB + j) * 16 + (lane & 15)) * BSTR], lane);
#pragma unroll
    for (int i = 0; i < 2; ++i)
      afr[i] = load_frag(&As[((wm * 2 + i) * 16 + (lane & 15)) * ASTR], lane);
#pragma unroll
    for (int i = 0; i < 2; ++i)
#pragma unroll
      for (int j = 0; j < NSUB; ++j)
        acc[i][j] = wmma_bf16(afr[i], bfr[j], acc[i][j]);
  }

  const int mhi = (lane >> 4) << 3;
#pragma unroll
  for (int i = 0; i < 2; ++i)
#pragma unroll
    for (int j = 0; j < NSUB; ++j) {
      int p = p0 + (wn * NSUB + j) * 16 + (lane & 15);
#pragma unroll
      for (int r = 0; r < 8; ++r) {
        int mg = m0 + (wm * 2 + i) * 16 + mhi + r;
        float v = acc[i][j][r];
        if (gq)   v = v * gq[mg] + bq[mg];
        if (bias) v += bias[mg];
        Yb[(size_t)mg * NSP + p] = act_apply(v, act);
      }
    }
}

// ---------------- KxK conv (pad=K/2, stride 1): implicit GEMM over taps.
// Border handling: clamp address + unconditional load + select (no EXEC branch).
template <int MT>
__global__ __launch_bounds__(256) void k_conv(
    const float* __restrict__ Wt, const float* __restrict__ X,
    float* __restrict__ Y, int Cout, int Cin, int H, int Wd, int KS,
    const float* __restrict__ gq, const float* __restrict__ bq, int act)
{
  constexpr int MW = MT / 32;
  constexpr int NW = 8 / MW;
  constexpr int NSUB = 64 / (NW * 16);
  __shared__ __align__(16) unsigned short As[MT * ASTR];
  __shared__ __align__(16) unsigned short Bs[64 * BSTR];
  const int tid  = threadIdx.x;
  const int lane = tid & 31, w = tid >> 5;
  const int wm = w / NW, wn = w % NW;
  const int HW = H * Wd;
  const int m0 = blockIdx.y * MT;
  const int p0 = blockIdx.x * 64;
  const int b  = blockIdx.z;
  const int KS2 = KS * KS, R = KS >> 1;
  const float* Xb = X + (size_t)b * Cin * HW;
  float* Yb = Y + (size_t)b * Cout * HW;

  v8f acc[2][NSUB];
#pragma unroll
  for (int i = 0; i < 2; ++i)
#pragma unroll
    for (int j = 0; j < NSUB; ++j) acc[i][j] = (v8f){0,0,0,0,0,0,0,0};

  for (int t = 0; t < KS2; ++t) {
    const int dh = t / KS - R, dw = t % KS - R;
    for (int k0 = 0; k0 < Cin; k0 += 32) {
      __syncthreads();
#pragma unroll
      for (int e = 0; e < MT / 8; ++e) {       // A: weights, stride KS2 (scalar)
        int idx = tid + e * 256;
        int m = idx >> 5, k = idx & 31;
        As[m * ASTR + k] =
            f2bfu(Wt[((size_t)(m0 + m) * Cin + k0 + k) * KS2 + t]);
      }
#pragma unroll
      for (int e = 0; e < 8; ++e) {            // B: shifted input, masked select
        int idx = tid + e * 256;
        int n = idx & 63, k = idx >> 6;
        int p = p0 + n;
        int hh = p / Wd + dh, ww = p % Wd + dw;
        bool inb = ((unsigned)hh < (unsigned)H) && ((unsigned)ww < (unsigned)Wd);
        int hc = min(max(hh, 0), H - 1), wc = min(max(ww, 0), Wd - 1);
        float v = Xb[(size_t)(k0 + k) * HW + hc * Wd + wc];
        Bs[n * BSTR + k] = f2bfu(inb ? v : 0.f);
      }
      __syncthreads();
      v16bf afr[2], bfr[NSUB];
#pragma unroll
      for (int j = 0; j < NSUB; ++j)
        bfr[j] = load_frag(&Bs[((wn * NSUB + j) * 16 + (lane & 15)) * BSTR], lane);
#pragma unroll
      for (int i = 0; i < 2; ++i)
        afr[i] = load_frag(&As[((wm * 2 + i) * 16 + (lane & 15)) * ASTR], lane);
#pragma unroll
      for (int i = 0; i < 2; ++i)
#pragma unroll
        for (int j = 0; j < NSUB; ++j)
          acc[i][j] = wmma_bf16(afr[i], bfr[j], acc[i][j]);
    }
  }

  const int mhi = (lane >> 4) << 3;
#pragma unroll
  for (int i = 0; i < 2; ++i)
#pragma unroll
    for (int j = 0; j < NSUB; ++j) {
      int p = p0 + (wn * NSUB + j) * 16 + (lane & 15);
#pragma unroll
      for (int r = 0; r < 8; ++r) {
        int mg = m0 + (wm * 2 + i) * 16 + mhi + r;
        float v = acc[i][j][r] * gq[mg] + bq[mg];
        Yb[(size_t)mg * HW + p] = act_apply(v, act);
      }
    }
}

// ---------------- LVC aggregation (NT GEMM, M=64, K contiguous both sides):
// En[b,k,c] = relu((sum_n A[b,k,n] z[b,c,n] - Asum[b,k] cw[k,c]) * ge[k] + be[k])
__global__ __launch_bounds__(256) void k_en_gemm(
    const float* __restrict__ Amat, const float* __restrict__ Z,
    const float* __restrict__ cw, const float* __restrict__ asum,
    const float* __restrict__ ge, const float* __restrict__ be,
    float* __restrict__ En, int NSP)
{
  __shared__ __align__(16) unsigned short As[64 * ASTR];
  __shared__ __align__(16) unsigned short Bs[64 * BSTR];
  const int tid  = threadIdx.x;
  const int lane = tid & 31, w = tid >> 5;
  const int wm = w >> 2, wn = w & 3;          // 2x4 wave grid, 32x16/wave
  const int n0 = blockIdx.x * 64;
  const int b  = blockIdx.z;
  const float* Ab = Amat + (size_t)b * 64 * NSP;
  const float* Zb = Z + (size_t)b * 256 * NSP;

  v8f acc0 = {0,0,0,0,0,0,0,0};
  v8f acc1 = acc0;

  for (int k0 = 0; k0 < NSP; k0 += 32) {
    __syncthreads();
#pragma unroll
    for (int e = 0; e < 2; ++e) {             // A: codes x pixels (K contiguous)
      int idx = tid + e * 256;
      int m = idx >> 3, k4 = (idx & 7) << 2;
      store_bf4(&As[m * ASTR + k4],
                *(const float4*)(Ab + (size_t)m * NSP + k0 + k4));
    }
#pragma unroll
    for (int e = 0; e < 2; ++e) {             // B: channels x pixels (K contiguous)
      int idx = tid + e * 256;
      int n = idx >> 3, k4 = (idx & 7) << 2;
      store_bf4(&Bs[n * BSTR + k4],
                *(const float4*)(Zb + (size_t)(n0 + n) * NSP + k0 + k4));
    }
    __syncthreads();
    v16bf bf = load_frag(&Bs[(wn * 16 + (lane & 15)) * BSTR], lane);
    v16bf a0 = load_frag(&As[(wm * 32 +      (lane & 15)) * ASTR], lane);
    v16bf a1 = load_frag(&As[(wm * 32 + 16 + (lane & 15)) * ASTR], lane);
    acc0 = wmma_bf16(a0, bf, acc0);
    acc1 = wmma_bf16(a1, bf, acc1);
  }

  const int ng  = n0 + wn * 16 + (lane & 15);
  const int mhi = (lane >> 4) << 3;
#pragma unroll
  for (int mt = 0; mt < 2; ++mt) {
    v8f acc = mt ? acc1 : acc0;
#pragma unroll
    for (int r = 0; r < 8; ++r) {
      int m = wm * 32 + mt * 16 + mhi + r;    // code index 0..63
      float v = acc[r] - asum[b * 64 + m] * cw[m * 256 + ng];
      v = v * ge[m] + be[m];
      En[((size_t)b * 64 + m) * 256 + ng] = fmaxf(v, 0.f);
    }
  }
}

// ---------------- small fp32 kernels ----------------
__global__ void k_maxpool(const float* __restrict__ X, float* __restrict__ Y,
                          int total, int H, int Wd) {
  int i = blockIdx.x * 256 + threadIdx.x;
  if (i >= total) return;
  int HW = H * Wd;
  int p = i % HW, bc = i / HW;
  int h = p / Wd, w = p % Wd;
  const float* s = X + (size_t)bc * HW;
  float m = -INFINITY;
  for (int dh = -1; dh <= 1; ++dh)
    for (int dw = -1; dw <= 1; ++dw) {
      int hh = h + dh, ww = w + dw;
      if (hh >= 0 && hh < H && ww >= 0 && ww < Wd)
        m = fmaxf(m, s[hh * Wd + ww]);
    }
  Y[i] = m;
}

__global__ void k_add_relu(const float* __restrict__ A, const float* __restrict__ B,
                           float* __restrict__ O, int n) {
  int i = blockIdx.x * 256 + threadIdx.x;
  if (i < n) O[i] = fmaxf(A[i] + B[i], 0.f);
}

__global__ void k_x2(const float* __restrict__ Z, float* __restrict__ X2o,
                     int Bn, int Cn, int NSP) {
  int i = blockIdx.x * 256 + threadIdx.x;
  if (i >= Bn * NSP) return;
  int b = i / NSP, n = i % NSP;
  const float* zb = Z + (size_t)b * Cn * NSP + n;
  float s = 0.f;
  for (int c = 0; c < Cn; ++c) { float v = zb[(size_t)c * NSP]; s += v * v; }
  X2o[i] = s;
}

__global__ void k_c2(const float* __restrict__ cw, float* __restrict__ c2,
                     int Kn, int Cn) {
  int k = threadIdx.x;
  if (k >= Kn) return;
  float s = 0.f;
  for (int c = 0; c < Cn; ++c) { float v = cw[k * Cn + c]; s += v * v; }
  c2[k] = s;
}

__global__ void k_softmax(float* __restrict__ A, const float* __restrict__ x2,
                          const float* __restrict__ c2, const float* __restrict__ sc,
                          int Bn, int Kn, int NSP) {
  int i = blockIdx.x * 256 + threadIdx.x;
  if (i >= Bn * NSP) return;
  int b = i / NSP, n = i % NSP;
  float xv = x2[i];
  float* col = A + (size_t)b * Kn * NSP + n;
  float mx = -INFINITY;
  for (int k = 0; k < Kn; ++k) {
    float d = sc[k] * (xv - 2.f * col[(size_t)k * NSP] + c2[k]);
    mx = fmaxf(mx, d);
  }
  float s = 0.f;
  for (int k = 0; k < Kn; ++k) {
    float d = sc[k] * (xv - 2.f * col[(size_t)k * NSP] + c2[k]);
    float e = expf(d - mx);
    col[(size_t)k * NSP] = e;
    s += e;
  }
  float inv = 1.f / s;
  for (int k = 0; k < Kn; ++k) col[(size_t)k * NSP] *= inv;
}

__global__ void k_rowsum(const float* __restrict__ A, float* __restrict__ out, int NSP) {
  __shared__ float red[256];
  const float* row = A + (size_t)blockIdx.x * NSP;
  float s = 0.f;
  for (int i = threadIdx.x; i < NSP; i += 256) s += row[i];
  red[threadIdx.x] = s; __syncthreads();
  for (int o = 128; o; o >>= 1) {
    if (threadIdx.x < o) red[threadIdx.x] += red[threadIdx.x + o];
    __syncthreads();
  }
  if (threadIdx.x == 0) out[blockIdx.x] = red[0];
}

__global__ void k_enmean(const float* __restrict__ En, float* __restrict__ Enm,
                         int Bn, int Kn, int Cn) {
  int i = blockIdx.x * 256 + threadIdx.x;
  if (i >= Bn * Cn) return;
  int b = i / Cn, c = i % Cn;
  float s = 0.f;
  for (int k = 0; k < Kn; ++k) s += En[((size_t)b * Kn + k) * Cn + c];
  Enm[i] = s / (float)Kn;
}

__global__ void k_gam(const float* __restrict__ Enm, const float* __restrict__ wfc,
                      const float* __restrict__ bfc, float* __restrict__ Gam,
                      int Bn, int Cn) {
  int i = blockIdx.x * 256 + threadIdx.x;
  if (i >= Bn * Cn) return;
  int b = i / Cn, co = i % Cn;
  const float* e = Enm + (size_t)b * Cn;
  const float* wr = wfc + (size_t)co * Cn;
  float s = bfc[co];
  for (int c = 0; c < Cn; ++c) s += e[c] * wr[c];
  Gam[i] = 1.f / (1.f + expf(-s));
}

__global__ void k_gate(const float* __restrict__ Xc, const float* __restrict__ Gam,
                       float* __restrict__ O, int Cn, int NSP, int total) {
  int i = blockIdx.x * 256 + threadIdx.x;
  if (i >= total) return;
  int c = (i / NSP) % Cn, b = i / (NSP * Cn);
  O[i] = fmaxf(Xc[i] * (1.f + Gam[b * Cn + c]), 0.f);
}

__global__ void k_gnstats(const float* __restrict__ X, float* __restrict__ mean,
                          float* __restrict__ rstd, int per) {
  __shared__ float s1[256], s2[256];
  const float* xb = X + (size_t)blockIdx.x * per;
  float a = 0.f, q = 0.f;
  for (int i = threadIdx.x; i < per; i += 256) { float v = xb[i]; a += v; q += v * v; }
  s1[threadIdx.x] = a; s2[threadIdx.x] = q; __syncthreads();
  for (int o = 128; o; o >>= 1) {
    if (threadIdx.x < o) { s1[threadIdx.x] += s1[threadIdx.x + o];
                           s2[threadIdx.x] += s2[threadIdx.x + o]; }
    __syncthreads();
  }
  if (threadIdx.x == 0) {
    float m = s1[0] / per;
    float var = s2[0] / per - m * m;
    mean[blockIdx.x] = m;
    rstd[blockIdx.x] = rsqrtf(var + 1e-5f);
  }
}

__global__ void k_gn_dw(const float* __restrict__ X, const float* __restrict__ mean,
                        const float* __restrict__ rstd, const float* __restrict__ g1,
                        const float* __restrict__ b1, const float* __restrict__ wdw,
                        const float* __restrict__ gdw, const float* __restrict__ bdw,
                        float* __restrict__ O, int Cn, int NSP, int total) {
  int i = blockIdx.x * 256 + threadIdx.x;
  if (i >= total) return;
  int c = (i / NSP) % Cn, b = i / (NSP * Cn);
  float v = (X[i] - mean[b]) * rstd[b] * g1[c] + b1[c];
  v = v * wdw[c];                 // depthwise 1x1 == per-channel scale
  v = v * gdw[c] + bdw[c];
  O[i] = v / (1.f + expf(-v));    // SiLU
}

__global__ void k_gn_apply(const float* __restrict__ X, const float* __restrict__ mean,
                           const float* __restrict__ rstd, const float* __restrict__ g1,
                           const float* __restrict__ b1, float* __restrict__ O,
                           int Cn, int NSP, int total) {
  int i = blockIdx.x * 256 + threadIdx.x;
  if (i >= total) return;
  int c = (i / NSP) % Cn, b = i / (NSP * Cn);
  O[i] = (X[i] - mean[b]) * rstd[b] * g1[c] + b1[c];
}

__global__ void k_axpy(const float* __restrict__ A, const float* __restrict__ S,
                       const float* __restrict__ X, float* __restrict__ O,
                       int Cn, int NSP, int total) {
  int i = blockIdx.x * 256 + threadIdx.x;
  if (i >= total) return;
  int c = (i / NSP) % Cn;
  O[i] = A[i] + S[c] * X[i];
}

// ---------------- host orchestration ----------------
extern "C" void kernel_launch(void* const* d_in, const int* in_sizes, int n_in,
                              void* d_out, int out_size, void* d_ws, size_t ws_size,
                              hipStream_t stream) {
  (void)in_sizes; (void)n_in; (void)out_size; (void)ws_size;
  const int B = 8, C = 256, H = 40, Wd = 40, HW = 1600, K = 64, HID = 1024;
  const int total = B * C * HW;                 // 3,276,800
  const size_t CH = (size_t)total;

  const float* x       = (const float*)d_in[0];
  const float* w_stem  = (const float*)d_in[1];
  const float* g_stem  = (const float*)d_in[2];
  const float* b_stem  = (const float*)d_in[3];
  const float* w_cb1   = (const float*)d_in[4];
  const float* g_cb1   = (const float*)d_in[5];
  const float* b_cb1   = (const float*)d_in[6];
  const float* w_cb2   = (const float*)d_in[7];
  const float* g_cb2   = (const float*)d_in[8];
  const float* b_cb2   = (const float*)d_in[9];
  const float* w_cb3   = (const float*)d_in[10];
  const float* g_cb3   = (const float*)d_in[11];
  const float* b_cb3   = (const float*)d_in[12];
  const float* w_cbres = (const float*)d_in[13];
  const float* g_cbres = (const float*)d_in[14];
  const float* b_cbres = (const float*)d_in[15];
  const float* w_lvc   = (const float*)d_in[16];
  const float* g_lvc   = (const float*)d_in[17];
  const float* b_lvc   = (const float*)d_in[18];
  const float* codew   = (const float*)d_in[19];
  const float* cw_sc   = (const float*)d_in[20];
  const float* g_enc   = (const float*)d_in[21];
  const float* b_enc   = (const float*)d_in[22];
  const float* w_fc    = (const float*)d_in[23];
  const float* b_fc    = (const float*)d_in[24];
  const float* gn1_g   = (const float*)d_in[25];
  const float* gn1_b   = (const float*)d_in[26];
  const float* w_dw    = (const float*)d_in[27];
  const float* g_dw    = (const float*)d_in[28];
  const float* b_dw    = (const float*)d_in[29];
  const float* w_pw    = (const float*)d_in[30];
  const float* g_pw    = (const float*)d_in[31];
  const float* b_pw    = (const float*)d_in[32];
  const float* gn2_g   = (const float*)d_in[33];
  const float* gn2_b   = (const float*)d_in[34];
  const float* w_fc1   = (const float*)d_in[35];
  const float* bias_fc1= (const float*)d_in[36];
  const float* w_fc2   = (const float*)d_in[37];
  const float* bias_fc2= (const float*)d_in[38];
  const float* ls1     = (const float*)d_in[39];
  const float* ls2     = (const float*)d_in[40];
  const float* w_cnv1  = (const float*)d_in[41];
  const float* b_cnv1  = (const float*)d_in[42];
  float* out = (float*)d_out;

  float* ws  = (float*)d_ws;
  float* bx1 = ws;                  // x1 (post-pool)
  float* bA  = ws + 1 * CH;         // scratch: stem pre-pool / h / gn2-out
  float* bXC = ws + 2 * CH;         // xc
  float* bZ  = ws + 3 * CH;         // z, later h2
  float* bLV = ws + 4 * CH;         // x_lvc
  float* bXM = ws + 5 * CH;         // xm
  float* bR  = ws + 6 * CH;         // residual / pw-out
  float* bQ1 = ws + 7 * CH;         // C/4 tensor
  float* bQ2 = bQ1 + CH / 4;        // C/4 tensor
  float* bHID= ws + 8 * CH;         // 4*CH hidden
  float* sm  = ws + 12 * CH;
  float* p_dot = sm;                                 // B*K*HW
  float* p_x2  = p_dot + (size_t)B * K * HW;         // B*HW
  float* p_c2  = p_x2 + (size_t)B * HW;              // K
  float* p_as  = p_c2 + K;                           // B*K
  float* p_en  = p_as + (size_t)B * K;               // B*K*C
  float* p_enm = p_en + (size_t)B * K * C;           // B*C
  float* p_gam = p_enm + (size_t)B * C;              // B*C
  float* p_gn  = p_gam + (size_t)B * C;              // 4*B stats

  dim3 blk(256);
  const int eb = (total + 255) / 256;                 // elementwise blocks
  const int pt = HW / 64;                             // pixel tiles = 25

  // ---- stem: conv7x7 + BN + ReLU, then maxpool ----
  k_conv<128><<<dim3(pt, C / 128, B), blk, 0, stream>>>(
      w_stem, x, bA, C, C, H, Wd, 7, g_stem, b_stem, ACT_RELU);
  k_maxpool<<<eb, blk, 0, stream>>>(bA, bx1, total, H, Wd);

  // ---- ConvBlock ----
  k_gemm<64><<<dim3(pt, 1, B), blk, 0, stream>>>(
      w_cb1, bx1, nullptr, C, bQ1, C / 4, C, HW, g_cb1, b_cb1, nullptr, ACT_RELU);
  k_conv<64><<<dim3(pt, 1, B), blk, 0, stream>>>(
      w_cb2, bQ1, bQ2, C / 4, C / 4, H, Wd, 3, g_cb2, b_cb2, ACT_RELU);
  k_gemm<128><<<dim3(pt, C / 128, B), blk, 0, stream>>>(
      w_cb3, bQ2, nullptr, C / 4, bXC, C, C / 4, HW, g_cb3, b_cb3, nullptr, ACT_NONE);
  k_gemm<128><<<dim3(pt, C / 128, B), blk, 0, stream>>>(
      w_cbres, bx1, nullptr, C, bR, C, C, HW, g_cbres, b_cbres, nullptr, ACT_NONE);
  k_add_relu<<<eb, blk, 0, stream>>>(bXC, bR, bXC, total);          // xc

  // ---- LVC ----
  k_gemm<128><<<dim3(pt, C / 128, B), blk, 0, stream>>>(
      w_lvc, bXC, nullptr, C, bZ, C, C, HW, g_lvc, b_lvc, nullptr, ACT_RELU);
  k_x2<<<(B * HW + 255) / 256, blk, 0, stream>>>(bZ, p_x2, B, C, HW);
  k_c2<<<1, 64, 0, stream>>>(codew, p_c2, K, C);
  k_gemm<64><<<dim3(pt, 1, B), blk, 0, stream>>>(
      codew, bZ, nullptr, C, p_dot, K, C, HW, nullptr, nullptr, nullptr, ACT_NONE);
  k_softmax<<<(B * HW + 255) / 256, blk, 0, stream>>>(p_dot, p_x2, p_c2, cw_sc, B, K, HW);
  k_rowsum<<<B * K, blk, 0, stream>>>(p_dot, p_as, HW);
  k_en_gemm<<<dim3(C / 64, 1, B), blk, 0, stream>>>(
      p_dot, bZ, codew, p_as, g_enc, b_enc, p_en, HW);
  k_enmean<<<(B * C + 255) / 256, blk, 0, stream>>>(p_en, p_enm, B, K, C);
  k_gam<<<(B * C + 255) / 256, blk, 0, stream>>>(p_enm, w_fc, b_fc, p_gam, B, C);
  k_gate<<<eb, blk, 0, stream>>>(bXC, p_gam, bLV, C, HW, total);    // x_lvc

  // ---- LightMLP ----
  k_gnstats<<<B, blk, 0, stream>>>(bx1, p_gn, p_gn + B, C * HW);
  k_gn_dw<<<eb, blk, 0, stream>>>(bx1, p_gn, p_gn + B, gn1_g, gn1_b,
                                  w_dw, g_dw, b_dw, bA, C, HW, total);
  k_gemm<128><<<dim3(pt, C / 128, B), blk, 0, stream>>>(
      w_pw, bA, nullptr, C, bR, C, C, HW, g_pw, b_pw, nullptr, ACT_SILU);
  k_axpy<<<eb, blk, 0, stream>>>(bx1, ls1, bR, bXM, C, HW, total);  // xm = x1 + ls1*h
  k_gnstats<<<B, blk, 0, stream>>>(bXM, p_gn + 2 * B, p_gn + 3 * B, C * HW);
  k_gn_apply<<<eb, blk, 0, stream>>>(bXM, p_gn + 2 * B, p_gn + 3 * B,
                                     gn2_g, gn2_b, bA, C, HW, total);
  k_gemm<128><<<dim3(pt, HID / 128, B), blk, 0, stream>>>(
      w_fc1, bA, nullptr, C, bHID, HID, C, HW, nullptr, nullptr, bias_fc1, ACT_GELU);
  k_gemm<128><<<dim3(pt, C / 128, B), blk, 0, stream>>>(
      w_fc2, bHID, nullptr, HID, bZ, C, HID, HW, nullptr, nullptr, bias_fc2, ACT_NONE);
  k_axpy<<<eb, blk, 0, stream>>>(bXM, ls2, bZ, bXM, C, HW, total);  // xm += ls2*h2

  // ---- fuse: 1x1 conv over concat([x_lvc, xm]) ----
  k_gemm<128><<<dim3(pt, C / 128, B), blk, 0, stream>>>(
      w_cnv1, bLV, bXM, C, out, C, 2 * C, HW, nullptr, nullptr, b_cnv1, ACT_NONE);
}
